// DelayGSCSNN_15831249453650
// MI455X (gfx1250) — matrix operand order repaired and checked by reading, not compiled, not run on hardware
//
#include <hip/hip_runtime.h>
#include <hip/hip_bf16.h>

typedef __attribute__((ext_vector_type(16))) _Float16 v16h;
typedef __attribute__((ext_vector_type(8)))  _Float16 v8h;
typedef __attribute__((ext_vector_type(8)))  float    v8f;
typedef __attribute__((ext_vector_type(2)))  float    v2f;

#define B_   256
#define T_   100
#define C_   40
#define H1_  1024
#define H2_  512
#define O_   35
#define OP_  48      // O padded to 3 wmma tiles
#define XS_  48      // x-stage row stride (floats)

#if defined(__has_builtin)
#if __has_builtin(__builtin_amdgcn_wmma_f32_16x16x4_f32)
#define HAS_WMMA_F32 1
#endif
#endif

// ---------------- fragment loaders ----------------
// A (16-bit, 16xK from LDS, row-major, stride K halves):
// lane L: M = L&15 ; 16 contiguous halves [kb+(L>>4)*16 .. +15]
__device__ __forceinline__ v16h ld_a16(const _Float16* p) {
  v8h lo = *(const v8h*)(p);
  v8h hi = *(const v8h*)(p + 8);
  v16h r;
#pragma unroll
  for (int i = 0; i < 8; ++i) { r[i] = lo[i]; r[i + 8] = hi[i]; }
  return r;
}
// B (16-bit, Kx16 = rows of W): lane L: N = L&15, 16 contiguous halves at
// K = kb + (L>>4)*16
__device__ __forceinline__ v16h ld_b16(const _Float16* p) {
  v8h lo = *(const v8h*)(p);
  v8h hi = *(const v8h*)(p + 8);
  v16h r;
#pragma unroll
  for (int i = 0; i < 8; ++i) { r[i] = lo[i]; r[i + 8] = hi[i]; }
  return r;
}
__device__ __forceinline__ v8f wmma16(v16h a, v16h b, v8f c) {
  return __builtin_amdgcn_wmma_f32_16x16x32_f16(false, a, false, b, (short)0, c, false, false);
}

// ---------------- prep kernels ----------------
__global__ void prep_bn(const float* __restrict__ gamma, const float* __restrict__ beta,
                        const float* __restrict__ mean,  const float* __restrict__ var,
                        float* __restrict__ g, float* __restrict__ b, int n) {
  int i = blockIdx.x * blockDim.x + threadIdx.x;
  if (i >= n) return;
  float gi = gamma[i] * rsqrtf(var[i] + 1e-5f);
  g[i] = gi;
  b[i] = beta[i] - mean[i] * gi;
}

__global__ void prep_delay(const float* __restrict__ raw, int* __restrict__ fli,
                           float* __restrict__ frac, int n) {
  int i = blockIdx.x * blockDim.x + threadIdx.x;
  if (i >= n) return;
  float d  = 30.0f / (1.0f + expf(-raw[i]));   // sigmoid * MAX_DELAY
  float fl = floorf(d);
  fli[i]  = (int)fl;
  frac[i] = d - fl;
}

__global__ void scale_rows_f32(const float* __restrict__ W, const float* __restrict__ g,
                               float* __restrict__ dst, int rows, int cols) {
  int i = blockIdx.x * blockDim.x + threadIdx.x;
  if (i >= rows * cols) return;
  dst[i] = W[i] * g[i / cols];
}

__global__ void conv_f16(const float* __restrict__ W, const float* __restrict__ g,
                         _Float16* __restrict__ dst, int rows, int cols, int src_rows) {
  int i = blockIdx.x * blockDim.x + threadIdx.x;
  if (i >= rows * cols) return;
  int r = i / cols;
  float v = 0.0f;
  if (r < src_rows) {
    v = W[(size_t)r * cols + (i % cols)];
    if (g) v *= g[r];
  }
  dst[i] = (_Float16)v;
}

// ---------------- main persistent SNN kernel ----------------
struct SnnParams {
  const float*    x;       // [B,T,C]
  const float*    Wd_s;    // [H1,C]  f32, BN-scaled
  const _Float16* Wr1;     // [H1,H1] f16, BN-scaled
  const _Float16* W2;      // [H2,H1] f16, BN-scaled
  const _Float16* Wr2;     // [H2,H2] f16, BN-scaled
  const _Float16* Wout;    // [OP,H2] f16 (rows >= O zero)
  const float*    b1;      // [H1]
  const float*    b2;      // [H2]
  const int*      fli;     // [C]
  const float*    frac;    // [C]
  const float*    alpha1; const float* rho1; const float* beta_a1;
  const float*    alpha2; const float* rho2; const float* beta_a2;
  const float*    beta_out;
  float*          out;     // [B,O]
};

__global__ __launch_bounds__(512, 1) void snn_persistent(SnnParams p) {
  __shared__ __align__(16) _Float16 s1_lds[16 * H1_];  // layer-1 spikes (f16 row-major)
  __shared__ __align__(16) _Float16 s2_lds[16 * H2_];  // layer-2 spikes
  __shared__ __align__(16) float    xs[16 * XS_];      // delayed input tile
  __shared__ int   fli_s[C_];
  __shared__ float frac_s[C_];

  const int tid   = threadIdx.x;
  const int lane  = tid & 31;
  const int wave  = tid >> 5;         // 0..15
  const int blk   = blockIdx.x;       // batch rows blk*16..+15
  const int n     = lane & 15;        // N within tile
  const int hsel  = lane >> 4;        // lane-half select
  const int khA   = hsel * 16;        // A fragment K sub-offset
  const int khB   = hsel * 16;        // B fragment K sub-offset
  const int m0    = hsel << 3;        // C/D row offset

  for (int i = tid; i < 16 * H1_; i += 512) s1_lds[i] = (_Float16)0.0f;
  for (int i = tid; i < 16 * H2_; i += 512) s2_lds[i] = (_Float16)0.0f;
  if (tid < C_) { fli_s[tid] = p.fli[tid]; frac_s[tid] = p.frac[tid]; }

  // persistent per-wave state (C/D layout: elem r -> M=r+m0, N=n)
  float v1[4][8], a1[4][8], v2[2][8], a2[2][8], vo[8], ac[8];
  float al1[4], rh1[4], ba1[4], bb1[4];
  float al2[2], rh2[2], ba2[2], bb2[2];
  float bon = 0.0f;

#pragma unroll
  for (int t0 = 0; t0 < 4; ++t0) {
    int j = wave * 64 + t0 * 16 + n;
    al1[t0] = p.alpha1[j]; rh1[t0] = p.rho1[j]; ba1[t0] = p.beta_a1[j]; bb1[t0] = p.b1[j];
#pragma unroll
    for (int r = 0; r < 8; ++r) { v1[t0][r] = 0.0f; a1[t0][r] = 0.0f; }
  }
#pragma unroll
  for (int t0 = 0; t0 < 2; ++t0) {
    int j = wave * 32 + t0 * 16 + n;
    al2[t0] = p.alpha2[j]; rh2[t0] = p.rho2[j]; ba2[t0] = p.beta_a2[j]; bb2[t0] = p.b2[j];
#pragma unroll
    for (int r = 0; r < 8; ++r) { v2[t0][r] = 0.0f; a2[t0][r] = 0.0f; }
  }
  if (wave < 3) {
    int o = wave * 16 + n;
    bon = (o < O_) ? p.beta_out[o] : 0.0f;
#pragma unroll
    for (int r = 0; r < 8; ++r) { vo[r] = 0.0f; ac[r] = 0.0f; }
  }

  // base pointers (tile offsets become compile-time immediates)
  const _Float16* arowS1 = s1_lds + n * H1_ + khA;
  const _Float16* arowS2 = s2_lds + n * H2_ + khA;
  const _Float16* browW1 = p.Wr1 + (size_t)(wave * 64 + n) * H1_ + khB;
  const _Float16* browW2 = p.W2  + (size_t)(wave * 32 + n) * H1_ + khB;
  const _Float16* browR2 = p.Wr2 + (size_t)(wave * 32 + n) * H2_ + khB;
  const _Float16* browWo = p.Wout + (size_t)(wave * 16 + n) * H2_ + khB;

  __syncthreads();

  for (int t = 0; t < T_; ++t) {
    // ---- stage fractionally-delayed input tile [16 x C] ----
    for (int i = tid; i < 16 * C_; i += 512) {
      int m = i / C_, c = i - m * C_;
      int b = blk * 16 + m;
      int i0 = t - fli_s[c];
      float fr = frac_s[c];
      float g0 = (i0 >= 0) ? p.x[((size_t)b * T_ + i0) * C_ + c] : 0.0f;
      float g1 = (i0 >= 1) ? p.x[((size_t)b * T_ + i0 - 1) * C_ + c] : 0.0f;
      xs[m * XS_ + c] = (1.0f - fr) * g0 + fr * g1;
    }
    __syncthreads();

    // ================= layer 1 : 4 N-tiles, shared A fragment =================
    {
      v8f acc[4];
#pragma unroll
      for (int t0 = 0; t0 < 4; ++t0) acc[t0] = (v8f){};

#if HAS_WMMA_F32
      { // input path: f32 WMMA 16x16x4 over K=C (A shared across tiles)
        const float* arowX = xs + n * XS_ + hsel * 2;
        const float* browD = p.Wd_s + (size_t)(wave * 64 + n) * C_ + hsel * 2;
#pragma unroll
        for (int kb = 0; kb < C_; kb += 4) {
          v2f av; av[0] = arowX[kb]; av[1] = arowX[kb + 1];
#pragma unroll
          for (int t0 = 0; t0 < 4; ++t0) {
            const float* bp = browD + t0 * (16 * C_) + kb;
            v2f bv; bv[0] = bp[0]; bv[1] = bp[1];
            acc[t0] = __builtin_amdgcn_wmma_f32_16x16x4_f32(false, av, false, bv,
                                                            (short)0, acc[t0], false, false);
          }
        }
      }
#else
      { // VALU fallback input path
        const float* browD = p.Wd_s + (size_t)(wave * 64 + n) * C_;
#pragma unroll
        for (int t0 = 0; t0 < 4; ++t0)
#pragma unroll
          for (int r = 0; r < 8; ++r) {
            float s = 0.0f;
            const float* xr = xs + (r + m0) * XS_;
            const float* wr = browD + t0 * (16 * C_);
            for (int c = 0; c < C_; ++c) s = fmaf(xr[c], wr[c], s);
            acc[t0][r] += s;
          }
      }
#endif
      // recurrent path: K=H1, one A load feeds 4 independent WMMAs
      for (int kbo = 0; kbo < H1_; kbo += 128) {
        __builtin_prefetch(browW1 + kbo + 128, 0, 0);   // warm next weight chunk
#pragma unroll
        for (int kbi = 0; kbi < 128; kbi += 32) {
          int kb = kbo + kbi;
          v16h a = ld_a16(arowS1 + kb);
#pragma unroll
          for (int t0 = 0; t0 < 4; ++t0)
            acc[t0] = wmma16(a, ld_b16(browW1 + t0 * (16 * H1_) + kb), acc[t0]);
        }
      }
      // adLIF update (BN shift folded into bias); spikes recomputed at publish
#pragma unroll
      for (int t0 = 0; t0 < 4; ++t0)
#pragma unroll
        for (int r = 0; r < 8; ++r) {
          float I  = acc[t0][r] + bb1[t0];
          float sp = (v1[t0][r] >= 1.0f) ? 1.0f : 0.0f;
          float an = rh1[t0] * a1[t0][r] + ba1[t0] * sp;
          float vn = al1[t0] * v1[t0][r] + (1.0f - al1[t0]) * I - an - sp;
          v1[t0][r] = vn; a1[t0][r] = an;
        }
    }
    __syncthreads();                 // all reads of old s1 done
#pragma unroll
    for (int t0 = 0; t0 < 4; ++t0)   // publish new layer-1 spikes
#pragma unroll
      for (int r = 0; r < 8; ++r)
        s1_lds[(r + m0) * H1_ + wave * 64 + t0 * 16 + n] =
            (_Float16)((v1[t0][r] >= 1.0f) ? 1.0f : 0.0f);
    __syncthreads();

    // ================= layer 2 : 2 N-tiles, shared A fragment =================
    {
      v8f acc[2];
#pragma unroll
      for (int t0 = 0; t0 < 2; ++t0) acc[t0] = (v8f){};
      for (int kbo = 0; kbo < H1_; kbo += 128) {       // s1 NEW @ W2
        __builtin_prefetch(browW2 + kbo + 128, 0, 0);
#pragma unroll
        for (int kbi = 0; kbi < 128; kbi += 32) {
          int kb = kbo + kbi;
          v16h a = ld_a16(arowS1 + kb);
#pragma unroll
          for (int t0 = 0; t0 < 2; ++t0)
            acc[t0] = wmma16(a, ld_b16(browW2 + t0 * (16 * H1_) + kb), acc[t0]);
        }
      }
      for (int kbo = 0; kbo < H2_; kbo += 128) {       // s2 OLD @ W_rec2
        __builtin_prefetch(browR2 + kbo + 128, 0, 0);
#pragma unroll
        for (int kbi = 0; kbi < 128; kbi += 32) {
          int kb = kbo + kbi;
          v16h a = ld_a16(arowS2 + kb);
#pragma unroll
          for (int t0 = 0; t0 < 2; ++t0)
            acc[t0] = wmma16(a, ld_b16(browR2 + t0 * (16 * H2_) + kb), acc[t0]);
        }
      }
#pragma unroll
      for (int t0 = 0; t0 < 2; ++t0)
#pragma unroll
        for (int r = 0; r < 8; ++r) {
          float I  = acc[t0][r] + bb2[t0];
          float sp = (v2[t0][r] >= 1.0f) ? 1.0f : 0.0f;
          float an = rh2[t0] * a2[t0][r] + ba2[t0] * sp;
          float vn = al2[t0] * v2[t0][r] + (1.0f - al2[t0]) * I - an - sp;
          v2[t0][r] = vn; a2[t0][r] = an;
        }
    }
    __syncthreads();                 // all reads of old s2 done
#pragma unroll
    for (int t0 = 0; t0 < 2; ++t0)   // publish new layer-2 spikes
#pragma unroll
      for (int r = 0; r < 8; ++r)
        s2_lds[(r + m0) * H2_ + wave * 32 + t0 * 16 + n] =
            (_Float16)((v2[t0][r] >= 1.0f) ? 1.0f : 0.0f);
    __syncthreads();

    // ================= leaky-integrator readout (waves 0..2) =================
    if (wave < 3) {
      v8f acc = {};
#pragma unroll 4
      for (int kb = 0; kb < H2_; kb += 32)             // s2 NEW @ W_out
        acc = wmma16(ld_a16(arowS2 + kb), ld_b16(browWo + kb), acc);
#pragma unroll
      for (int r = 0; r < 8; ++r) {
        float von = bon * vo[r] + (1.0f - bon) * acc[r];
        vo[r] = von; ac[r] += von;
      }
    }
    // readout's s2 reads complete before next iteration's s2-write barrier.
  }

  if (wave < 3) {
    int ob = wave * 16;
#pragma unroll
    for (int r = 0; r < 8; ++r) {
      int b = blk * 16 + r + m0;
      int o = ob + n;
      if (o < O_) p.out[(size_t)b * O_ + o] = ac[r] * (1.0f / (float)T_);
    }
  }
}

// ---------------- host launcher ----------------
extern "C" void kernel_launch(void* const* d_in, const int* in_sizes, int n_in,
                              void* d_out, int out_size, void* d_ws, size_t ws_size,
                              hipStream_t stream) {
  const float* x         = (const float*)d_in[0];
  const float* W_delay   = (const float*)d_in[1];
  const float* delay_raw = (const float*)d_in[2];
  const float* W_rec1    = (const float*)d_in[3];
  const float* W2        = (const float*)d_in[4];
  const float* W_rec2    = (const float*)d_in[5];
  const float* W_out     = (const float*)d_in[6];
  const float* bn1_g = (const float*)d_in[7],  *bn1_b = (const float*)d_in[8];
  const float* bn1_m = (const float*)d_in[9],  *bn1_v = (const float*)d_in[10];
  const float* bn2_g = (const float*)d_in[11], *bn2_b = (const float*)d_in[12];
  const float* bn2_m = (const float*)d_in[13], *bn2_v = (const float*)d_in[14];
  const float* alpha1 = (const float*)d_in[15], *rho1 = (const float*)d_in[16], *beta_a1 = (const float*)d_in[17];
  const float* alpha2 = (const float*)d_in[18], *rho2 = (const float*)d_in[19], *beta_a2 = (const float*)d_in[20];
  const float* beta_out = (const float*)d_in[21];

  char*  base = (char*)d_ws;
  size_t off  = 0;
  auto carve = [&](size_t bytes) -> char* {
    char* ptr = base + off;
    off = (off + bytes + 255) & ~(size_t)255;
    return ptr;
  };
  float*     g1    = (float*)carve(H1_ * 4);
  float*     b1    = (float*)carve(H1_ * 4);
  float*     g2    = (float*)carve(H2_ * 4);
  float*     b2    = (float*)carve(H2_ * 4);
  int*       fli   = (int*)carve(C_ * 4);
  float*     frac  = (float*)carve(C_ * 4);
  float*     Wd_s  = (float*)carve((size_t)H1_ * C_ * 4);
  _Float16*  Wr1h  = (_Float16*)carve((size_t)H1_ * H1_ * 2);
  _Float16*  W2h   = (_Float16*)carve((size_t)H2_ * H1_ * 2);
  _Float16*  Wr2h  = (_Float16*)carve((size_t)H2_ * H2_ * 2);
  _Float16*  Wouth = (_Float16*)carve((size_t)OP_ * H2_ * 2);

  prep_bn<<<(H1_ + 255) / 256, 256, 0, stream>>>(bn1_g, bn1_b, bn1_m, bn1_v, g1, b1, H1_);
  prep_bn<<<(H2_ + 255) / 256, 256, 0, stream>>>(bn2_g, bn2_b, bn2_m, bn2_v, g2, b2, H2_);
  prep_delay<<<1, 64, 0, stream>>>(delay_raw, fli, frac, C_);
  {
    int nWd = H1_ * C_;
    scale_rows_f32<<<(nWd + 255) / 256, 256, 0, stream>>>(W_delay, g1, Wd_s, H1_, C_);
    int n1 = H1_ * H1_;
    conv_f16<<<(n1 + 255) / 256, 256, 0, stream>>>(W_rec1, g1, Wr1h, H1_, H1_, H1_);
    int n2 = H2_ * H1_;
    conv_f16<<<(n2 + 255) / 256, 256, 0, stream>>>(W2, g2, W2h, H2_, H1_, H2_);
    int n3 = H2_ * H2_;
    conv_f16<<<(n3 + 255) / 256, 256, 0, stream>>>(W_rec2, g2, Wr2h, H2_, H2_, H2_);
    int n4 = OP_ * H2_;
    conv_f16<<<(n4 + 255) / 256, 256, 0, stream>>>(W_out, nullptr, Wouth, OP_, H2_, O_);
  }

  SnnParams prm;
  prm.x = x; prm.Wd_s = Wd_s; prm.Wr1 = Wr1h; prm.W2 = W2h; prm.Wr2 = Wr2h; prm.Wout = Wouth;
  prm.b1 = b1; prm.b2 = b2; prm.fli = fli; prm.frac = frac;
  prm.alpha1 = alpha1; prm.rho1 = rho1; prm.beta_a1 = beta_a1;
  prm.alpha2 = alpha2; prm.rho2 = rho2; prm.beta_a2 = beta_a2;
  prm.beta_out = beta_out;
  prm.out = (float*)d_out;

  snn_persistent<<<16, 512, 0, stream>>>(prm);
}
